// DFFN_21723944583182
// MI455X (gfx1250) — matrix operand order repaired and checked by reading, not compiled, not run on hardware
//
#include <hip/hip_runtime.h>
#include <hip/hip_bf16.h>
#include <math.h>

// ---------------------------------------------------------------------------
// MI455X (gfx1250) spectral-FFN block.
//   prep_w   : swizzle w_in/w_before/w_out -> WMMA A-fragment order (bf16)
//   prep_T   : real rfft2 filter == 8x8 circular conv -> circulant 64x64 T_c,
//              swizzled to A-fragment order (bf16)
//   gemm1    : Y = w_in @ X   (B frags direct from global fp32)   -> yb planar
//   fftconv  : Z = T_c @ patches via global_load_async_to_lds_b128 staging,
//              gelu-gate -> gbuf pixel-major [b][pix][64]
//   gemm2    : G2 = w_before @ G (B frags direct), D restaged in LDS and
//              written out with global_store_async_from_lds_b128
//   dw       : depthwise 3x3 SAME, channel-vectorized (pixel-major)
//   gemm3    : OUT = w_out @ H3 + X (fp32 out, planar)
// ---------------------------------------------------------------------------

typedef __attribute__((ext_vector_type(16))) __bf16 v16bf;
typedef __attribute__((ext_vector_type(8)))  float  v8f;

#define BSZ   8
#define CF    128
#define CH    64
#define HDIM  256
#define WDIM  256
#define HW    65536

__device__ __constant__ float c_cos8[8] =
    {1.0f, 0.70710678f, 0.0f, -0.70710678f, -1.0f, -0.70710678f, 0.0f, 0.70710678f};

// --------- WMMA fragment index maps (CDNA5 ISA 7.12.2, wave32) -------------
// A (16x32 bf16): lane = m + 16*g ; element e -> K
__device__ __forceinline__ int a_k_of(int e, int g) {
    return 16 * g + e;
}
// B (32x16 bf16): lane = n + 16*g ; element e -> K = 16*g + e (contiguous)

__device__ __forceinline__ v8f wmma_bf16(v16bf a, v16bf b, v8f c) {
    return __builtin_amdgcn_wmma_f32_16x16x32_bf16(
        false, a, false, b, (short)0, c, false, false);
}

// --------- CDNA5 async LDS<->memory (ASYNCcnt-tracked) ---------------------
// GVS mode: mem = SADDR(sgpr64) + VADDR(u32).
// LDS operand: low 32 bits of the generic address of the __shared__ object
// (aperture mapping, ISA 10.2); deriving it from the
// real pointer also *captures* the LDS array so the "memory" clobber keeps
// the surrounding LDS stores/loads alive (DCE-safe).
__device__ __forceinline__ void async_copy16(void* lds_dst,
                                             const void* sbase,
                                             unsigned g_byte_off) {
    unsigned loff = (unsigned)(size_t)lds_dst;
    asm volatile("global_load_async_to_lds_b128 %0, %1, %2"
                 :: "v"(loff), "v"(g_byte_off), "s"(sbase)
                 : "memory");
}
__device__ __forceinline__ void async_store16(unsigned g_byte_off,
                                              const void* lds_src,
                                              const void* sbase) {
    unsigned loff = (unsigned)(size_t)lds_src;
    asm volatile("global_store_async_from_lds_b128 %0, %1, %2"
                 :: "v"(g_byte_off), "v"(loff), "s"(sbase)
                 : "memory");
}
__device__ __forceinline__ void wait_async0() {
    asm volatile("s_wait_asynccnt 0" ::: "memory");
}

// ---------------------------------------------------------------------------
// prep: swizzle row-major MxK fp32 -> A-fragment order bf16:
// dst[((mt*ktiles + kt)*32 + lane)*16 + e]
// ---------------------------------------------------------------------------
__device__ void swizzle_mat(const float* __restrict__ src, int M, int K,
                            __bf16* __restrict__ dst) {
    const int ktiles = K >> 5;
    const int total  = (M >> 4) * ktiles * 512;
    for (int idx = threadIdx.x; idx < total; idx += blockDim.x) {
        int tile = idx >> 9, rr = idx & 511, lane = rr >> 4, e = rr & 15;
        int mt = tile / ktiles, kt = tile % ktiles;
        int g = lane >> 4, mr = lane & 15;
        int m = mt * 16 + mr;
        int k = kt * 32 + a_k_of(e, g);
        dst[idx] = (__bf16)src[m * K + k];
    }
}

__global__ void k_prep_w(const float* __restrict__ w_in,
                         const float* __restrict__ w_before,
                         const float* __restrict__ w_out,
                         __bf16* __restrict__ awin,
                         __bf16* __restrict__ abef,
                         __bf16* __restrict__ aout) {
    swizzle_mat(w_in,     CF, CF, awin);
    swizzle_mat(w_before, CH, CH, abef);
    swizzle_mat(w_out,    CF, CH, aout);
}

// ---------------------------------------------------------------------------
// prep_T: kc[d] = (1/64) sum_k Ffull[k] cos(2pi<k,d>/8); T_c circulant of kc.
// ---------------------------------------------------------------------------
__global__ void k_prep_T(const float* __restrict__ fft_filt,
                         __bf16* __restrict__ tws) {
    const int c = blockIdx.x;
    __shared__ float Ff[64];
    __shared__ float kc[64];
    const int tid = threadIdx.x;
    if (tid < 64) {
        int k1 = tid >> 3, k2 = tid & 7;
        float v;
        if (k2 <= 4) v = fft_filt[c * 40 + k1 * 5 + k2];
        else         v = fft_filt[c * 40 + (((8 - k1) & 7) * 5) + (8 - k2)];
        Ff[tid] = v;
    }
    __syncthreads();
    if (tid < 64) {
        int d1 = tid >> 3, d2 = tid & 7;
        float s = 0.f;
        #pragma unroll
        for (int k1 = 0; k1 < 8; ++k1)
            #pragma unroll
            for (int k2 = 0; k2 < 8; ++k2)
                s += Ff[k1 * 8 + k2] * c_cos8[(k1 * d1 + k2 * d2) & 7];
        kc[tid] = s * (1.0f / 64.0f);
    }
    __syncthreads();
    for (int idx = tid; idx < 4096; idx += 256) {
        int tile = idx >> 9, rr = idx & 511, lane = rr >> 4, e = rr & 15;
        int mt = tile >> 1, kt = tile & 1;
        int g = lane >> 4, mr = lane & 15;
        int m = mt * 16 + mr;
        int k = kt * 32 + a_k_of(e, g);
        int d1 = ((m >> 3) - (k >> 3)) & 7;
        int d2 = ((m & 7) - (k & 7)) & 7;
        tws[c * 4096 + idx] = (__bf16)kc[d1 * 8 + d2];
    }
}

// ---------------------------------------------------------------------------
// gemm1: Y(128 x HW) = w_in @ X per batch, bf16 planar out.
// B-fragment elements are per-e 64B-coalesced global fp32 loads (no LDS).
// ---------------------------------------------------------------------------
__global__ __launch_bounds__(256)
void k_gemm1(const float* __restrict__ x, const __bf16* __restrict__ awin,
             __bf16* __restrict__ yb) {
    const int blk = blockIdx.x;
    const int bb  = blk >> 9;
    const int po  = (blk & 511) << 7;
    const int tid = threadIdx.x;
    const int wave = tid >> 5, lane = tid & 31;
    const int g = lane >> 4, nn = lane & 15;
    const float* xb = x + (size_t)bb * CF * HW + po + wave * 16 + nn;
    __builtin_prefetch(awin, 0, 0);
    v8f acc[8] = {};
    #pragma unroll
    for (int kt = 0; kt < 4; ++kt) {
        v16bf bf;
        #pragma unroll
        for (int e = 0; e < 16; ++e)
            bf[e] = (__bf16)xb[(size_t)(kt * 32 + 16 * g + e) * HW];
        #pragma unroll
        for (int mt = 0; mt < 8; ++mt) {
            v16bf af = *(const v16bf*)(awin + (size_t)(((mt * 4 + kt) * 32) + lane) * 16);
            acc[mt] = wmma_bf16(af, bf, acc[mt]);
        }
    }
    __bf16* yo = yb + (size_t)bb * CF * HW + po + wave * 16 + nn;
    #pragma unroll
    for (int mt = 0; mt < 8; ++mt)
        #pragma unroll
        for (int r = 0; r < 8; ++r) {
            int co = mt * 16 + r + 8 * g;
            yo[(size_t)co * HW] = (__bf16)acc[mt][r];
        }
}

// ---------------------------------------------------------------------------
// fftconv: per (b, c, band of 128 patches): stage both channels to LDS in
// patch-major order via global_load_async_to_lds_b128; Z = T @ P (wmma);
// gelu-gate; write gbuf pixel-major [b][pix][64].
// ---------------------------------------------------------------------------
__global__ __launch_bounds__(256)
void k_fftconv(const __bf16* __restrict__ yb, const __bf16* __restrict__ tws,
               __bf16* __restrict__ gbuf) {
    const int blk  = blockIdx.x;
    const int band = blk & 7;
    const int c    = (blk >> 3) & 63;
    const int bb   = blk >> 9;
    __shared__ __bf16 tile[2 * 128 * 64];    // [half][patch][pixel], 32KB
    const int rowbase = band * 32;
    const int tid = threadIdx.x;

    // 2048 16-byte segments: one 8-pixel patch-row-segment each.
    for (int i = tid; i < 2048; i += 256) {
        int half = i >> 10, seg = i & 1023;
        int row = seg >> 5, pc = seg & 31;
        int p   = (row >> 3) * 32 + pc;
        int sub = row & 7;
        unsigned goff = (unsigned)((((bb * CF + c + half * 64) * HW)
                                    + (rowbase + row) * WDIM + pc * 8) * 2);
        async_copy16(&tile[half * 8192 + p * 64 + sub * 8], yb, goff);
    }
    wait_async0();
    __syncthreads();

    const int wave = tid >> 5, lane = tid & 31;
    const int g = lane >> 4, nn = lane & 15;
    const int p = wave * 16 + nn;
    const int pr = p >> 5, pc = p & 31;
    v8f z1[4] = {}, z2[4] = {};
    const __bf16* t0 = tws + (size_t)c * 4096;
    const __bf16* t1 = tws + (size_t)(c + 64) * 4096;
    #pragma unroll
    for (int kt = 0; kt < 2; ++kt) {
        v16bf bf0 = *(const v16bf*)&tile[p * 64 + kt * 32 + 16 * g];
        v16bf bf1 = *(const v16bf*)&tile[8192 + p * 64 + kt * 32 + 16 * g];
        #pragma unroll
        for (int mt = 0; mt < 4; ++mt) {
            v16bf a0 = *(const v16bf*)(t0 + (size_t)(((mt * 2 + kt) * 32) + lane) * 16);
            v16bf a1 = *(const v16bf*)(t1 + (size_t)(((mt * 2 + kt) * 32) + lane) * 16);
            z1[mt] = wmma_bf16(a0, bf0, z1[mt]);
            z2[mt] = wmma_bf16(a1, bf1, z2[mt]);
        }
    }
    __bf16* go = gbuf + (size_t)bb * HW * CH;
    #pragma unroll
    for (int mt = 0; mt < 4; ++mt)
        #pragma unroll
        for (int r = 0; r < 8; ++r) {
            int m = mt * 16 + r + 8 * g;
            float a1v = z1[mt][r], a2v = z2[mt][r];
            float gl = 0.5f * a1v * (1.0f + erff(a1v * 0.70710678f));
            int row = rowbase + pr * 8 + (m >> 3);
            int col = pc * 8 + (m & 7);
            go[((size_t)row * WDIM + col) * CH + c] = (__bf16)(gl * a2v);
        }
}

// ---------------------------------------------------------------------------
// gemm2: G2 = w_before @ G, pixel-major in/out. B-frags direct from global;
// D restaged via LDS, written out with global_store_async_from_lds_b128.
// ---------------------------------------------------------------------------
__global__ __launch_bounds__(256)
void k_gemm2(const __bf16* __restrict__ gbuf, const __bf16* __restrict__ abef,
             __bf16* __restrict__ g2) {
    __shared__ __bf16 dsr[128 * 64];         // [pix_local][ch], 16KB
    const int blk = blockIdx.x;
    const int bb  = blk >> 9;
    const int po  = (blk & 511) << 7;
    const int tid = threadIdx.x;
    const int wave = tid >> 5, lane = tid & 31;
    const int g = lane >> 4, nn = lane & 15;
    const int pxl = wave * 16 + nn;
    const __bf16* gb = gbuf + ((size_t)bb * HW + po + pxl) * CH;
    v8f acc[4] = {};
    #pragma unroll
    for (int kt = 0; kt < 2; ++kt) {
        v16bf bf = *(const v16bf*)(gb + kt * 32 + 16 * g);
        #pragma unroll
        for (int mt = 0; mt < 4; ++mt) {
            v16bf af = *(const v16bf*)(abef + (size_t)(((mt * 2 + kt) * 32) + lane) * 16);
            acc[mt] = wmma_bf16(af, bf, acc[mt]);
        }
    }
    #pragma unroll
    for (int mt = 0; mt < 4; ++mt)
        #pragma unroll
        for (int r = 0; r < 8; ++r) {
            int co = mt * 16 + r + 8 * g;
            dsr[pxl * CH + co] = (__bf16)acc[mt][r];
        }
    __syncthreads();
    // async DMA the 16KB tile LDS -> global (coalesced 16B segments)
    const unsigned gbase = (unsigned)(((size_t)bb * HW + po) * CH * 2);
    for (int i = tid; i < 1024; i += 256)
        async_store16(gbase + (unsigned)(i * 16), &dsr[i * 8], g2);
    wait_async0();
}

// ---------------------------------------------------------------------------
// dw: depthwise 3x3 SAME on pixel-major [b][pix][64] bf16.
// One block per (b,row); each thread one pixel, 4 chunks of 16 channels.
// ---------------------------------------------------------------------------
__global__ __launch_bounds__(256)
void k_dw(const __bf16* __restrict__ g2, const float* __restrict__ wdw,
          __bf16* __restrict__ h3) {
    const int blk = blockIdx.x;
    const int row = blk & 255;
    const int bb  = blk >> 8;
    __shared__ float ws9[9 * CH];            // [tap][ch]
    const int tid = threadIdx.x;
    for (int i = tid; i < 9 * CH; i += 256) {
        int ch = i / 9, tap = i % 9;
        ws9[tap * CH + ch] = wdw[i];
    }
    __syncthreads();
    const int xcol = tid;
    const __bf16* sbase = g2 + (size_t)bb * HW * CH;
    __bf16*       dbase = h3 + ((size_t)bb * HW + (size_t)row * WDIM + xcol) * CH;
    #pragma unroll
    for (int chunk = 0; chunk < 4; ++chunk) {
        float acc[16];
        #pragma unroll
        for (int t = 0; t < 16; ++t) acc[t] = 0.f;
        #pragma unroll
        for (int di = -1; di <= 1; ++di) {
            int y = row + di;
            if (y < 0 || y >= HDIM) continue;
            #pragma unroll
            for (int dj = -1; dj <= 1; ++dj) {
                int xc = xcol + dj;
                if (xc < 0 || xc >= WDIM) continue;
                int tap = (di + 1) * 3 + (dj + 1);
                const __bf16* sp = sbase + ((size_t)y * WDIM + xc) * CH + chunk * 16;
                #pragma unroll
                for (int t = 0; t < 16; ++t)
                    acc[t] += ws9[tap * CH + chunk * 16 + t] * (float)sp[t];
            }
        }
        #pragma unroll
        for (int t = 0; t < 16; ++t)
            dbase[chunk * 16 + t] = (__bf16)acc[t];
    }
}

// ---------------------------------------------------------------------------
// gemm3: OUT = w_out @ H3 + X, fp32 planar out. B-frags direct from global.
// ---------------------------------------------------------------------------
__global__ __launch_bounds__(256)
void k_gemm3(const __bf16* __restrict__ h3, const __bf16* __restrict__ aout,
             const float* __restrict__ x, float* __restrict__ out) {
    const int blk = blockIdx.x;
    const int bb  = blk >> 9;
    const int po  = (blk & 511) << 7;
    const int tid = threadIdx.x;
    const int wave = tid >> 5, lane = tid & 31;
    const int g = lane >> 4, nn = lane & 15;
    const int pxl = wave * 16 + nn;
    const __bf16* hb = h3 + ((size_t)bb * HW + po + pxl) * CH;
    v8f acc[8] = {};
    #pragma unroll
    for (int kt = 0; kt < 2; ++kt) {
        v16bf bf = *(const v16bf*)(hb + kt * 32 + 16 * g);
        #pragma unroll
        for (int mt = 0; mt < 8; ++mt) {
            v16bf af = *(const v16bf*)(aout + (size_t)(((mt * 2 + kt) * 32) + lane) * 16);
            acc[mt] = wmma_bf16(af, bf, acc[mt]);
        }
    }
    const float* xr = x   + (size_t)bb * CF * HW + po + pxl;
    float*       oo = out + (size_t)bb * CF * HW + po + pxl;
    #pragma unroll
    for (int mt = 0; mt < 8; ++mt)
        #pragma unroll
        for (int r = 0; r < 8; ++r) {
            int co = mt * 16 + r + 8 * g;
            oo[(size_t)co * HW] = acc[mt][r] + xr[(size_t)co * HW];
        }
}

// ---------------------------------------------------------------------------
extern "C" void kernel_launch(void* const* d_in, const int* in_sizes, int n_in,
                              void* d_out, int out_size, void* d_ws, size_t ws_size,
                              hipStream_t stream) {
    const float* x        = (const float*)d_in[0];
    const float* fft_filt = (const float*)d_in[1];
    const float* w_in     = (const float*)d_in[2];
    const float* w_before = (const float*)d_in[3];
    const float* w_dw     = (const float*)d_in[4];
    const float* w_out    = (const float*)d_in[5];
    float* out = (float*)d_out;

    char* ws = (char*)d_ws;
    __bf16* awin = (__bf16*)(ws);                       //  32 KB
    __bf16* abef = (__bf16*)(ws + 32768);               //   8 KB
    __bf16* aout = (__bf16*)(ws + 40960);               //  16 KB
    __bf16* tws  = (__bf16*)(ws + 65536);               //   1 MB
    const size_t big = 2u << 20;
    __bf16* yb   = (__bf16*)(ws + big);                 // 128 MB planar (B,128,HW)
    __bf16* gbuf = (__bf16*)(ws + big + 134217728ull);  //  64 MB pixel-major (B,HW,64)
    __bf16* g2   = yb;                                  // reuse (yb dead after fftconv)
    __bf16* h3   = gbuf;                                // reuse (gbuf dead after gemm2)

    k_prep_w <<<1,    256, 0, stream>>>(w_in, w_before, w_out, awin, abef, aout);
    k_prep_T <<<128,  256, 0, stream>>>(fft_filt, tws);
    k_gemm1  <<<4096, 256, 0, stream>>>(x, awin, yb);
    k_fftconv<<<4096, 256, 0, stream>>>(yb, tws, gbuf);
    k_gemm2  <<<4096, 256, 0, stream>>>(gbuf, abef, g2);
    k_dw     <<<2048, 256, 0, stream>>>(g2, w_dw, h3);
    k_gemm3  <<<4096, 256, 0, stream>>>(h3, aout, x, out);
}
